// DINAttention_85091892068522
// MI455X (gfx1250) — matrix operand order repaired and checked by reading, not compiled, not run on hardware
//
#include <hip/hip_runtime.h>

typedef __attribute__((ext_vector_type(16))) _Float16 v16h;
typedef __attribute__((ext_vector_type(8)))  float    v8f;

#define L_SEQ 200
#define LP    208          // L padded to 13 tiles of 16
#define DDIM  64
#define CIN   256          // 4*D concat features
#define H1N   80
#define H1P   96           // H1 padded K for GEMM2 (3 x 32)
#define H2N   40
#define H2P   48           // H2 padded N (3 x 16)
#define W1TILES (5 * 8)    // nt x kt tiles of W1 fragments
#define W2TILES (3 * 3)
#define NEG_INF_F (-4294967295.0f)   // -2^32 + 1, matches reference

// ---- prepass: pack weights into WMMA B-fragment layout (f16) ---------------
// Fragment tile = 512 halfs: lane-major [lane][e], e = 2*v+j, so that each
// lane's 16 halfs are one contiguous 32-byte v16h load in the main kernel.
// B layout (32x16 f16, wave32): lanes 0-15 hold K=0..7,16..23, lanes 16-31 K=8..15,24..31.
__global__ __launch_bounds__(256) void din_pack_weights(
    const float* __restrict__ W1, const float* __restrict__ W2,
    _Float16* __restrict__ w1p, _Float16* __restrict__ w2p) {
  int idx = blockIdx.x * 256 + threadIdx.x;
  if (idx < W1TILES * 512) {
    int tile = idx >> 9, r = idx & 511;
    int lane = r >> 4, e = r & 15;
    int nt = tile >> 3, kt = tile & 7;           // tile = nt*8 + kt
    int k = kt * 32 + ((lane < 16) ? 0 : 8) + ((e < 8) ? e : (8 + e));
    int n = nt * 16 + (lane & 15);
    w1p[idx] = (_Float16)W1[k * H1N + n];
  }
  if (idx < W2TILES * 512) {
    int tile = idx >> 9, r = idx & 511;
    int lane = r >> 4, e = r & 15;
    int nt = tile / 3, kt = tile % 3;            // tile = nt*3 + kt
    int k = kt * 32 + ((lane < 16) ? 0 : 8) + ((e < 8) ? e : (8 + e));
    int n = nt * 16 + (lane & 15);
    float v = (k < H1N && n < H2N) ? W2[k * H2N + n] : 0.0f;
    w2p[idx] = (_Float16)v;
  }
}

// ---- main fused DIN attention kernel: one block per batch row --------------
__global__ __launch_bounds__(256) void din_attention_kernel(
    const float* __restrict__ query, const float* __restrict__ keys,
    const float* __restrict__ values, const int* __restrict__ mask,
    const float* __restrict__ b1, const float* __restrict__ a1,
    const float* __restrict__ b2, const float* __restrict__ a2,
    const float* __restrict__ Wo, const float* __restrict__ bo,
    const _Float16* __restrict__ w1p, const _Float16* __restrict__ w2p,
    float* __restrict__ out) {
  __shared__ _Float16 sK[LP * DDIM];                       // 26.0 KB
  __shared__ _Float16 sQ[DDIM];
  __shared__ alignas(32) _Float16 sW1p[W1TILES * 512];     // 40.0 KB packed frags
  __shared__ alignas(32) _Float16 sW2p[W2TILES * 512];     // 9.0 KB packed frags
  __shared__ _Float16 sH1[LP * H1P];                       // 39.0 KB, cols 80..95 zero
  __shared__ float    sB1[H1N], sA1[H1N];
  __shared__ float    sB2[H2N], sA2[H2N], sWo[H2N];
  __shared__ float    sScore[LP];
  __shared__ float    sRed[16];
  __shared__ float    sAcc[256];

  const int b    = blockIdx.x;
  const int tid  = threadIdx.x;
  const int lane = tid & 31;
  const int wave = tid >> 5;

  const float* vp = values + (size_t)b * L_SEQ * DDIM;
  for (int i = tid * 32; i < L_SEQ * DDIM; i += 256 * 32)
    __builtin_prefetch(vp + i, 0, 1);

  // ---------------- Phase 0: stage into LDS ----------------
  if (tid < DDIM) sQ[tid] = (_Float16)query[(size_t)b * DDIM + tid];
  {
    const float4* k4 = (const float4*)(keys + (size_t)b * L_SEQ * DDIM);
    for (int i = tid; i < (L_SEQ * DDIM) / 4; i += 256) {
      float4 v = k4[i];
      sK[4 * i + 0] = (_Float16)v.x;
      sK[4 * i + 1] = (_Float16)v.y;
      sK[4 * i + 2] = (_Float16)v.z;
      sK[4 * i + 3] = (_Float16)v.w;
    }
  }
  for (int i = tid; i < (LP - L_SEQ) * DDIM; i += 256)
    sK[L_SEQ * DDIM + i] = (_Float16)0.0f;
  for (int i = tid; i < LP * (H1P - H1N); i += 256) {     // zero pad cols of H1
    int r = i / (H1P - H1N), c = H1N + (i % (H1P - H1N));
    sH1[r * H1P + c] = (_Float16)0.0f;
  }
  {
    const uint4* s1 = (const uint4*)w1p; uint4* d1 = (uint4*)sW1p;
    for (int i = tid; i < (W1TILES * 512) / 8; i += 256) d1[i] = s1[i];
    const uint4* s2 = (const uint4*)w2p; uint4* d2 = (uint4*)sW2p;
    for (int i = tid; i < (W2TILES * 512) / 8; i += 256) d2[i] = s2[i];
  }
  if (tid < H1N) { sB1[tid] = b1[tid]; sA1[tid] = a1[tid]; }
  if (tid < H2N) { sB2[tid] = b2[tid]; sA2[tid] = a2[tid]; sWo[tid] = Wo[tid]; }
  {
    float bias0 = bo[0];
    for (int i = tid; i < LP; i += 256) sScore[i] = bias0;
  }
  __syncthreads();

  // fragment layout helpers (wave32, per ISA 7.12.2)
  const int nl    = lane & 15;
  const int mhalf = (lane < 16) ? 0 : 8;    // C/D row offset
  const int akb   = (lane < 16) ? 0 : 8;    // A K-base

  // q-only A-fragments (feature cols 0..63 are q[k], row-independent): hoisted
  v16h aQ0, aQ1;
#pragma unroll
  for (int v = 0; v < 8; ++v) {
    int base = (v < 4) ? (akb + 2 * v) : (16 + akb + 2 * (v - 4));
    aQ0[2 * v] = sQ[base];      aQ0[2 * v + 1] = sQ[base + 1];
    aQ1[2 * v] = sQ[32 + base]; aQ1[2 * v + 1] = sQ[33 + base];
  }

  // ---------------- GEMM1: X[208x256] @ W1[256x80] -> H1 ----------------
  for (int t = wave; t < 13 * 5; t += 8) {
    int mt = t / 5, nt = t % 5;
    int m = mt * 16 + nl;
    int n = nt * 16 + nl;
    v8f acc = {};
#pragma unroll
    for (int kt = 0; kt < 8; ++kt) {
      v16h af;
      if (kt == 0)      af = aQ0;
      else if (kt == 1) af = aQ1;
      else {
#pragma unroll
        for (int v = 0; v < 8; ++v) {
          int base = (v < 4) ? (akb + 2 * v) : (16 + akb + 2 * (v - 4));
          int kk = ((kt & 1) << 5) + base;
          _Float16 k0 = sK[m * DDIM + kk], k1 = sK[m * DDIM + kk + 1];
          _Float16 x0, x1;
          if (kt < 4)      { x0 = k0; x1 = k1; }                      // k
          else {
            _Float16 q0 = sQ[kk], q1 = sQ[kk + 1];
            if (kt < 6)    { x0 = q0 - k0; x1 = q1 - k1; }            // q-k
            else           { x0 = q0 * k0; x1 = q1 * k1; }            // q*k
          }
          af[2 * v] = x0; af[2 * v + 1] = x1;
        }
      }
      const v16h bf = *(const v16h*)(sW1p + ((((nt << 3) + kt) << 5) + lane) * 16);
      acc = __builtin_amdgcn_wmma_f32_16x16x32_f16(false, af, false, bf,
                                                   (short)0, acc, false, false);
    }
    float bias = sB1[n], alpha = sA1[n];
#pragma unroll
    for (int v = 0; v < 8; ++v) {
      int mr = mt * 16 + v + mhalf;
      float x = acc[v] + bias;
      x = fmaxf(x, 0.0f) + alpha * fminf(x, 0.0f);
      sH1[mr * H1P + n] = (_Float16)x;
    }
  }
  __syncthreads();

  // ------- GEMM2: H1[208x96] @ W2[96x48] ; fused PReLU + (.)@Wo -> score ----
  for (int t = wave; t < 13 * 3; t += 8) {
    int mt = t / 3, nt = t % 3;
    int m = mt * 16 + nl;
    int n = nt * 16 + nl;
    v8f acc = {};
#pragma unroll
    for (int kt = 0; kt < 3; ++kt) {
      v16h af;
#pragma unroll
      for (int v = 0; v < 8; ++v) {
        int ka = kt * 32 + ((v < 4) ? (akb + 2 * v) : (16 + akb + 2 * (v - 4)));
        af[2 * v]     = sH1[m * H1P + ka];
        af[2 * v + 1] = sH1[m * H1P + ka + 1];
      }
      const v16h bf = *(const v16h*)(sW2p + (((nt * 3 + kt) << 5) + lane) * 16);
      acc = __builtin_amdgcn_wmma_f32_16x16x32_f16(false, af, false, bf,
                                                   (short)0, acc, false, false);
    }
    if (n < H2N) {
      float bias = sB2[n], alpha = sA2[n], wo = sWo[n];
#pragma unroll
      for (int v = 0; v < 8; ++v) {
        int mr = mt * 16 + v + mhalf;
        float x = acc[v] + bias;
        x = fmaxf(x, 0.0f) + alpha * fminf(x, 0.0f);
        atomicAdd(&sScore[mr], x * wo);     // ds_add_f32: fused H2 @ Wo
      }
    }
  }
  __syncthreads();

  // ---------------- mask + softmax over L ----------------
  float s = -3.0e38f;
  if (tid < L_SEQ) {
    s = sScore[tid];
    if (mask[(size_t)b * L_SEQ + tid] == 0) s = NEG_INF_F;
  }
  float r = s;
#pragma unroll
  for (int off = 16; off > 0; off >>= 1) r = fmaxf(r, __shfl_xor(r, off, 32));
  if (lane == 0) sRed[wave] = r;
  __syncthreads();
  float gmax = sRed[0];
#pragma unroll
  for (int i = 1; i < 8; ++i) gmax = fmaxf(gmax, sRed[i]);
  float e = (tid < L_SEQ) ? __expf(s - gmax) : 0.0f;
  float rs = e;
#pragma unroll
  for (int off = 16; off > 0; off >>= 1) rs += __shfl_xor(rs, off, 32);
  if (lane == 0) sRed[8 + wave] = rs;
  __syncthreads();
  float gsum = sRed[8];
#pragma unroll
  for (int i = 1; i < 8; ++i) gsum += sRed[8 + i];
  if (tid < L_SEQ) sScore[tid] = e / gsum;
  __syncthreads();

  // ---------------- out[b] = probs @ values[b] ----------------
  const int d = tid & 63;
  const int g = tid >> 6;
  float acc = 0.0f;
  for (int l = g; l < L_SEQ; l += 4)
    acc += sScore[l] * vp[l * DDIM + d];
  sAcc[tid] = acc;
  __syncthreads();
  if (g == 0)
    out[(size_t)b * DDIM + d] = sAcc[d] + sAcc[64 + d] + sAcc[128 + d] + sAcc[192 + d];
}

extern "C" void kernel_launch(void* const* d_in, const int* in_sizes, int n_in,
                              void* d_out, int out_size, void* d_ws, size_t ws_size,
                              hipStream_t stream) {
  const float* query  = (const float*)d_in[0];
  const float* keys   = (const float*)d_in[1];
  const float* values = (const float*)d_in[2];
  const int*   mask   = (const int*)d_in[3];
  const float* W1 = (const float*)d_in[4];
  const float* b1 = (const float*)d_in[5];
  const float* a1 = (const float*)d_in[6];
  const float* W2 = (const float*)d_in[7];
  const float* b2 = (const float*)d_in[8];
  const float* a2 = (const float*)d_in[9];
  const float* Wo = (const float*)d_in[10];
  const float* bo = (const float*)d_in[11];
  float* out = (float*)d_out;

  _Float16* w1p = (_Float16*)d_ws;                 // 40*512 halfs
  _Float16* w2p = w1p + W1TILES * 512;             // 9*512 halfs

  int B = in_sizes[0] / DDIM;   // 2048

  din_pack_weights<<<(W1TILES * 512 + 255) / 256, 256, 0, stream>>>(W1, W2, w1p, w2p);
  din_attention_kernel<<<B, 256, 0, stream>>>(query, keys, values, mask,
                                              b1, a1, b2, a2, Wo, bo,
                                              w1p, w2p, out);
}
